// KnnLookupLayer_90933047591274
// MI455X (gfx1250) — matrix operand order repaired and checked by reading, not compiled, not run on hardware
//
#include <hip/hip_runtime.h>
#include <stdint.h>

#define B_Q   512
#define N_DB  131072
#define D_DIM 512
#define K_NN  32
#define KITER (D_DIM / 32)

typedef unsigned short u16;
typedef __attribute__((ext_vector_type(16))) __bf16       v16bf;
typedef __attribute__((ext_vector_type(8)))  float        v8f;
typedef __attribute__((ext_vector_type(4)))  unsigned int u32x4;
typedef __attribute__((ext_vector_type(4)))  int          i32x4;
typedef __attribute__((ext_vector_type(8)))  int          i32x8;
typedef __attribute__((ext_vector_type(4)))  float        f32x4;

union Frag16 { u32x4 u[2]; v16bf v; };

__device__ __forceinline__ u16 f32_to_bf16(float f) {
  union { float f; unsigned u; } x; x.f = f;
  unsigned r = x.u + 0x7FFFu + ((x.u >> 16) & 1u);  // round-to-nearest-even
  return (u16)(r >> 16);
}

// ---------------------------------------------------------------------------
// Kernel 1: queries f32 -> bf16
// ---------------------------------------------------------------------------
__global__ __launch_bounds__(256) void conv_q(const float* __restrict__ q,
                                              u16* __restrict__ qh) {
  int i = blockIdx.x * 256 + threadIdx.x;
  if (i < B_Q * D_DIM) qh[i] = f32_to_bf16(q[i]);
}

// ---------------------------------------------------------------------------
// Kernel 2: database f32 -> bf16 + row sum-of-squares (one block per row)
// ---------------------------------------------------------------------------
__global__ __launch_bounds__(128) void prep_db(const float* __restrict__ db,
                                               u16* __restrict__ dbh,
                                               float* __restrict__ dbsq) {
  __shared__ float red[128];
  const int row = blockIdx.x;
  const int t   = threadIdx.x;
  const float* src = db  + (size_t)row * D_DIM;
  u16*         dst = dbh + (size_t)row * D_DIM;
  float s = 0.f;
  for (int i = t; i < D_DIM; i += 128) {
    float v = src[i];
    s += v * v;
    dst[i] = f32_to_bf16(v);
  }
  red[t] = s;
  __syncthreads();
  for (int off = 64; off > 0; off >>= 1) {
    if (t < off) red[t] += red[t + off];
    __syncthreads();
  }
  if (t == 0) dbsq[row] = red[0];
}

// ---------------------------------------------------------------------------
// Device helper: issue one TDM 2D tile load (256 db rows x 32 bf16) into LDS.
// ---------------------------------------------------------------------------
__device__ __forceinline__ void tdm_load_tile(unsigned long long gaddr,
                                              unsigned lds_addr) {
  u32x4 g0;
  g0[0] = 1u;                                   // count=1, user descriptor
  g0[1] = lds_addr;                             // lds_addr (bytes)
  g0[2] = (unsigned)(gaddr & 0xFFFFFFFFull);    // global_addr[31:0]
  g0[3] = (unsigned)((gaddr >> 32) & 0x1FFFFFFull) | (2u << 30);  // type=2
  i32x8 g1;
  g1[0] = (int)(1u << 16);                      // data_size=1 -> 2 bytes
  g1[1] = (int)((D_DIM & 0xFFFFu) << 16);       // tensor_dim0 lo16
  g1[2] = (int)((D_DIM >> 16) | ((N_DB & 0xFFFF) << 16));  // dim0 hi | dim1 lo
  g1[3] = (int)(((unsigned)N_DB >> 16) | (32u << 16));     // dim1 hi | tile_dim0=32
  g1[4] = (int)256;                             // tile_dim1=256, tile_dim2=0
  g1[5] = (int)D_DIM;                           // tensor_dim0_stride lo32
  g1[6] = 0;
  g1[7] = 0;
  i32x4 z4 = {};
#if defined(__clang_major__) && __clang_major__ >= 23
  i32x8 z8 = {};
  __builtin_amdgcn_tensor_load_to_lds(g0, g1, z4, z4, z8, 0);
#else
  __builtin_amdgcn_tensor_load_to_lds(g0, g1, z4, z4, 0);
#endif
}

// ---------------------------------------------------------------------------
// Kernel 3: WMMA bf16 GEMM -> scores[b,n] = db_sq[n] - 2*dot(q_b, db_n)
// Block: 256 threads (8 waves). Tile: 128 rows x 256 cols. K-chunks of 32.
// B tiles double-buffered in LDS via the Tensor Data Mover (transfer of chunk
// k+1 overlaps the 16 WMMAs of chunk k; tensor ops of a wave complete in
// order, so s_wait_tensorcnt 1 after issuing the next transfer guarantees the
// current buffer is ready). A fragments and B fragments software-pipelined.
// ---------------------------------------------------------------------------
__global__ __launch_bounds__(256) void knn_gemm(const u16* __restrict__ qh,
                                                const u16* __restrict__ dbh,
                                                const float* __restrict__ dbsq,
                                                float* __restrict__ scores) {
  __shared__ __align__(16) u16   smemB[2][256 * 32];  // 2 x 16 KB, col c at c*64B
  __shared__ __align__(16) float smemSq[256];

  const int tid  = threadIdx.x;
  const int wave = tid >> 5;
  const int lane = tid & 31;
  const int ln   = lane & 15;
  const int hi   = (lane >> 4) & 1;

  const int n0     = blockIdx.x * 256;
  const int m_base = blockIdx.y * 128 + wave * 16;

  // Stage db_sq[n0 .. n0+255] into LDS with async global->LDS loads (waves 0,1).
  if (tid < 64) {
    unsigned lds_off = (unsigned)(uintptr_t)(&smemSq[tid * 4]);
    unsigned goff    = (unsigned)((n0 + tid * 4) * sizeof(float));
    unsigned long long sbase = (unsigned long long)(uintptr_t)dbsq;
    asm volatile("global_load_async_to_lds_b128 %0, %1, %2"
                 :: "v"(lds_off), "v"(goff), "s"(sbase) : "memory");
  }

  v8f acc[16];
  v8f vzero = {};
#pragma unroll
  for (int j = 0; j < 16; ++j) acc[j] = vzero;

  const unsigned long long dbh_base =
      (unsigned long long)(uintptr_t)dbh + (unsigned long long)n0 * D_DIM * 2ull;
  const unsigned smemB_lds = (unsigned)(uintptr_t)(&smemB[0][0]);
  const u16* arow_base = qh + (size_t)(m_base + ln) * D_DIM;

  // Prologue: TDM for chunk 0 into buffer 0; preload A fragment for chunk 0.
  if (wave == 0) {
    tdm_load_tile(dbh_base, smemB_lds);
  }
  Frag16 acur;
  {
    const u32x4* ap = (const u32x4*)arow_base;
    acur.u[0] = ap[hi];
    acur.u[1] = ap[hi + 2];
  }

  for (int kk = 0; kk < KITER; ++kk) {
    const int k0 = kk * 32;

    // Issue TDM for chunk kk+1 into the other buffer, then wait until chunk kk
    // (the older outstanding transfer) has landed.
    if (wave == 0) {
      if (kk + 1 < KITER) {
        tdm_load_tile(dbh_base + (unsigned long long)(k0 + 32) * 2ull,
                      smemB_lds + (unsigned)(((kk + 1) & 1) * (256 * 32 * 2)));
        __builtin_amdgcn_s_wait_tensorcnt(1);
      } else {
        __builtin_amdgcn_s_wait_tensorcnt(0);
      }
    }
    __syncthreads();

    // Preload next A fragment (hides global latency under the WMMA chain).
    Frag16 anext = acur;
    if (kk + 1 < KITER) {
      const u32x4* ap = (const u32x4*)(arow_base + k0 + 32);
      anext.u[0] = ap[hi];
      anext.u[1] = ap[hi + 2];
      __builtin_prefetch(arow_base + k0 + 64, 0, 1);
    }

    const u16* bufc = &smemB[kk & 1][0];

    // Software-pipelined B fragments: load j+1 before issuing WMMA j.
    Frag16 bcur;
    {
      const u32x4* bp = (const u32x4*)(bufc + (size_t)ln * 32 + hi * 8);
      bcur.u[0] = bp[0];
      bcur.u[1] = bp[2];
    }
#pragma unroll
    for (int j = 0; j < 16; ++j) {
      Frag16 bnext = bcur;
      if (j + 1 < 16) {
        const u32x4* bp =
            (const u32x4*)(bufc + (size_t)((j + 1) * 16 + ln) * 32 + hi * 8);
        bnext.u[0] = bp[0];
        bnext.u[1] = bp[2];
      }
      acc[j] = __builtin_amdgcn_wmma_f32_16x16x32_bf16(
          false, acur.v, false, bcur.v, (short)0, acc[j], false, false);
      bcur = bnext;
    }
    acur = anext;
    __syncthreads();  // all reads of buf[kk&1] done before TDM reuses it
  }

  if (tid < 64) {
    asm volatile("s_wait_asynccnt 0" ::: "memory");
  }
  __syncthreads();

  // C layout: lane ln = column, VGPR r -> row r + hi*8.
#pragma unroll
  for (int j = 0; j < 16; ++j) {
    int n = n0 + j * 16 + ln;
    float sq = smemSq[j * 16 + ln];
#pragma unroll
    for (int r = 0; r < 8; ++r) {
      int m = m_base + hi * 8 + r;
      scores[(size_t)m * N_DB + n] = sq - 2.0f * acc[j][r];
    }
  }
}

// ---------------------------------------------------------------------------
// Kernel 4: per-row top-32 (ascending distance) + gather of db rows.
// One 256-thread block per query. Per-thread 32-entry max-heap in LDS
// (layout [j*256+t] -> bank-conflict-free), then 32 arg-min extractions.
// ---------------------------------------------------------------------------
__global__ __launch_bounds__(256) void topk_gather(const float* __restrict__ scores,
                                                   const float* __restrict__ db,
                                                   float* __restrict__ out) {
  __shared__ float hval[32 * 256];
  __shared__ int   hidx[32 * 256];
  __shared__ float rmin[256];
  __shared__ int   rpos[256];
  __shared__ int   resi[32];

  const int b = blockIdx.x;
  const int t = threadIdx.x;
  const float* srow = scores + (size_t)b * N_DB;
  const float FINF = 3.402823466e38f;

  int cnt = 0;
  for (int n = t; n < N_DB; n += 256) {
    float v = srow[n];
    if (cnt < 32) {
      hval[cnt * 256 + t] = v;
      hidx[cnt * 256 + t] = n;
      ++cnt;
      if (cnt == 32) {  // heapify (max-heap of current 32 smallest)
        for (int s = 15; s >= 0; --s) {
          int j = s;
          for (;;) {
            int l = 2 * j + 1, r = 2 * j + 2, m = j;
            if (l < 32 && hval[l * 256 + t] > hval[m * 256 + t]) m = l;
            if (r < 32 && hval[r * 256 + t] > hval[m * 256 + t]) m = r;
            if (m == j) break;
            float tv = hval[j * 256 + t]; hval[j * 256 + t] = hval[m * 256 + t]; hval[m * 256 + t] = tv;
            int   ti = hidx[j * 256 + t]; hidx[j * 256 + t] = hidx[m * 256 + t]; hidx[m * 256 + t] = ti;
            j = m;
          }
        }
      }
    } else if (v < hval[t]) {  // replace root, sift down
      hval[t] = v;
      hidx[t] = n;
      int j = 0;
      for (;;) {
        int l = 2 * j + 1, r = 2 * j + 2, m = j;
        if (l < 32 && hval[l * 256 + t] > hval[m * 256 + t]) m = l;
        if (r < 32 && hval[r * 256 + t] > hval[m * 256 + t]) m = r;
        if (m == j) break;
        float tv = hval[j * 256 + t]; hval[j * 256 + t] = hval[m * 256 + t]; hval[m * 256 + t] = tv;
        int   ti = hidx[j * 256 + t]; hidx[j * 256 + t] = hidx[m * 256 + t]; hidx[m * 256 + t] = ti;
        j = m;
      }
    }
  }
  __syncthreads();

  // 32 rounds of block-wide arg-min over the 8192 candidates (ascending order).
  for (int s = 0; s < 32; ++s) {
    float lv = FINF;
    int   lp = t;
    for (int j = 0; j < 32; ++j) {
      float v = hval[j * 256 + t];
      if (v < lv) { lv = v; lp = j * 256 + t; }
    }
    rmin[t] = lv;
    rpos[t] = lp;
    __syncthreads();
    for (int off = 128; off > 0; off >>= 1) {
      if (t < off && rmin[t + off] < rmin[t]) {
        rmin[t] = rmin[t + off];
        rpos[t] = rpos[t + off];
      }
      __syncthreads();
    }
    if (t == 0) {
      int p = rpos[0];
      resi[s] = hidx[p];
      hval[p] = FINF;  // remove winner
    }
    __syncthreads();
  }

  // Gather: out[b, r, :] = database[resi[r], :]  (f32, 128 float4 per row)
  for (int r = 0; r < K_NN; ++r) {
    int row = resi[r];
    const f32x4* src = (const f32x4*)(db + (size_t)row * D_DIM);
    f32x4*       dst = (f32x4*)(out + ((size_t)b * K_NN + r) * D_DIM);
    if (t < 128) dst[t] = src[t];
  }
}

// ---------------------------------------------------------------------------
// Host-side launch
// ---------------------------------------------------------------------------
extern "C" void kernel_launch(void* const* d_in, const int* in_sizes, int n_in,
                              void* d_out, int out_size, void* d_ws, size_t ws_size,
                              hipStream_t stream) {
  (void)in_sizes; (void)n_in; (void)out_size; (void)ws_size;

  const float* d_q  = (const float*)d_in[0];   // [B, D] f32
  const float* d_db = (const float*)d_in[1];   // [N, D] f32
  float*       out  = (float*)d_out;           // [B, K, D] f32

  char* ws = (char*)d_ws;
  const size_t off_qh     = 0;
  const size_t off_dbh    = off_qh  + (size_t)B_Q  * D_DIM * sizeof(u16);   // 512 KB
  const size_t off_dbsq   = off_dbh + (size_t)N_DB * D_DIM * sizeof(u16);   // +128 MB
  const size_t off_scores = off_dbsq + (size_t)N_DB * sizeof(float);        // +512 KB

  u16*   qh     = (u16*)(ws + off_qh);
  u16*   dbh    = (u16*)(ws + off_dbh);
  float* dbsq   = (float*)(ws + off_dbsq);
  float* scores = (float*)(ws + off_scores);

  conv_q<<<(B_Q * D_DIM + 255) / 256, 256, 0, stream>>>(d_q, qh);
  prep_db<<<N_DB, 128, 0, stream>>>(d_db, dbh, dbsq);
  knn_gemm<<<dim3(N_DB / 256, B_Q / 128), 256, 0, stream>>>(qh, dbh, dbsq, scores);
  topk_gather<<<B_Q, 256, 0, stream>>>(scores, d_db, out);
}
